// RGB_SFMCNN_V2_58351425683923
// MI455X (gfx1250) — matrix operand order repaired and checked by reading, not compiled, not run on hardware
//
#include <hip/hip_runtime.h>

// ---------------- problem constants ----------------
#define IN_CH   64
#define OUT_CH  256
#define HW      56          // input H == W
#define HO      54          // output H == W
#define P       (HO * HO)   // 2916 pixels per image
#define KDIM    576         // IN_CH * 3 * 3
#define BATCH   16

#define HTILE   8           // output rows per block
#define WTILE   16          // output cols per block
#define MTILE   (HTILE * WTILE)   // 128 pixels per block
#define NTILE   128               // out channels per block
#define PITCHA  72          // LDS A row pitch (bf16): 64 data + 8 pad (TDM pad)
#define PITCHO  132         // LDS out-stage pitch in floats
#define ABUF_BYTES (MTILE * PITCHA * 2)   // 18432 per buffer

typedef __attribute__((ext_vector_type(16))) __bf16        v16bf;
typedef __attribute__((ext_vector_type(8)))  float         v8f;
typedef __attribute__((ext_vector_type(4)))  unsigned int  uint4v;
typedef unsigned int u32x4 __attribute__((ext_vector_type(4)));
typedef int          i32x8 __attribute__((ext_vector_type(8)));
typedef int          i32x4 __attribute__((ext_vector_type(4)));

union Frag16 { uint4v q[2]; v16bf v; };

__device__ __forceinline__ unsigned short f32_to_bf16_rne(float f) {
    unsigned int u = __float_as_uint(f);
    u += 0x7FFFu + ((u >> 16) & 1u);
    return (unsigned short)(u >> 16);
}

// A fragment (16x32 bf16) from LDS row-major [m][k], pitch PITCHA.
__device__ __forceinline__ v16bf load_a_frag(const unsigned short* ldsA,
                                             int mrow, int k0, int lane) {
    const int m    = mrow + (lane & 15);
    const int half = lane >> 4;
    const unsigned short* pb = ldsA + m * PITCHA + k0 + half * 8;
    Frag16 f;
    f.q[0] = *(const uint4v*)(pb);        // K = base .. base+7
    f.q[1] = *(const uint4v*)(pb + 16);   // K = base+16 .. base+23
    return f.v;
}

// B fragment (32x16 bf16) straight from K-contiguous weights [o][k'].
__device__ __forceinline__ v16bf load_b_frag(const unsigned short* wB,
                                             int nbase, int kp, int lane) {
    const unsigned short* pb =
        wB + (size_t)(nbase + (lane & 15)) * KDIM + kp + (lane >> 4) * 8;
    Frag16 f;
    f.q[0] = *(const uint4v*)(pb);
    f.q[1] = *(const uint4v*)(pb + 16);
    return f.v;
}

// ---- TDM: DMA one (c=64) x (w=16) x (h=8) bf16 tile of xT into LDS,
// padded to PITCHA per row; OOB rows/cols zero-filled by the engine.
__device__ __forceinline__ void tdm_load_a(const unsigned short* xTb,
                                           unsigned lds_off, int ho, int wo) {
    const unsigned remW = (unsigned)(HW - wo);    // remaining w extent
    const unsigned remH = (unsigned)(HW - ho);    // remaining h extent
    const unsigned long long ga =
        (unsigned long long)(size_t)(xTb + ((size_t)ho * HW + wo) * IN_CH);

    u32x4 g0;
    g0[0] = 1u;                                   // count = 1 (valid D#)
    g0[1] = lds_off;                              // LDS byte address
    g0[2] = (unsigned)ga;                         // global_addr[31:0]
    g0[3] = (unsigned)(ga >> 32) | (2u << 30);    // global_addr[56:32] | type=2

    i32x8 g1;
    g1[0] = (1 << 16)          // data_size = 2B
          | (1 << 20)          // pad_enable
          | (4 << 22)          // pad_interval: every 32 DWORDs (=128B row)
          | (3 << 25);         // pad_amount: 4 DWORDs (=8 bf16) -> pitch 72
    g1[1] = (int)((IN_CH & 0xFFFFu) << 16);                 // tensor_dim0 lo16
    g1[2] = (int)((remW & 0xFFFFu) << 16);                  // dim0 hi=0 | dim1 lo16
    g1[3] = (int)((remW >> 16) | (64u << 16));              // dim1 hi | tile_dim0=64
    g1[4] = (int)(16u | (8u << 16));                        // tile_dim1=16, tile_dim2=8
    g1[5] = (int)IN_CH;                                     // dim0_stride = 64
    g1[6] = (int)((3584u & 0xFFFFu) << 16);                 // stride0 hi | dim1_stride lo16 (56*64)
    g1[7] = (int)(3584u >> 16);                             // dim1_stride hi

    i32x4 g2;
    g2[0] = (int)remH;    // tensor_dim2
    g2[1] = 1;            // tensor_dim3 (unused)
    g2[2] = 0;            // tensor_dim2_stride (unused, tile_dim3=0)
    g2[3] = 0;            // tile_dim3 = 0
    i32x4 g3; g3[0] = 0; g3[1] = 0; g3[2] = 0; g3[3] = 0;
    i32x8 gx; gx[0] = 0; gx[1] = 0; gx[2] = 0; gx[3] = 0;
    gx[4] = 0; gx[5] = 0; gx[6] = 0; gx[7] = 0;

    __builtin_amdgcn_tensor_load_to_lds(g0, g1, g2, g3, gx, 0);
}

// ---------------- pre-pass A: xT[b][h][w][c] = bf16(x[b][c][h][w]),
// sq[b][h][w] = sum_c x^2 (f32) ----------------
__global__ __launch_bounds__(256)
void rbf_xt_kernel(const float* __restrict__ x,
                   unsigned int* __restrict__ xT,     // bf16 pairs
                   float* __restrict__ sq) {
    __shared__ unsigned int ldsT[256 * 33];            // pitch 33 for banks
    const int t = threadIdx.x;
    const int pixBase = blockIdx.x * 256;              // 196 * 256 == 50176 exact
    const int pix = pixBase + t;
    const int b = pix / (HW * HW);
    const int rem = pix - b * (HW * HW);

    const float* xp = x + (size_t)b * IN_CH * HW * HW + rem;
    float ss = 0.f;
#pragma unroll
    for (int c = 0; c < IN_CH; c += 2) {
        float v0 = xp[(size_t)c * (HW * HW)];
        float v1 = xp[(size_t)(c + 1) * (HW * HW)];
        ss += v0 * v0 + v1 * v1;
        ldsT[t * 33 + (c >> 1)] =
            (unsigned)f32_to_bf16_rne(v0) | ((unsigned)f32_to_bf16_rne(v1) << 16);
    }
    sq[pix] = ss;
    __syncthreads();
    // contiguous b128 writes: 4 uints per thread per iter
#pragma unroll
    for (int i = 0; i < 8; ++i) {
        const int idx = (i * 256 + t) * 4;     // uint index in tile
        const int pl  = idx >> 5;              // pixel local
        const int cw  = idx & 31;              // uint within 32-uint row
        uint4v v;
        v[0] = ldsT[pl * 33 + cw];
        v[1] = ldsT[pl * 33 + cw + 1];
        v[2] = ldsT[pl * 33 + cw + 2];
        v[3] = ldsT[pl * 33 + cw + 3];
        *(uint4v*)(xT + (((size_t)(pixBase + pl)) << 5) + cw) = v;
    }
}

// ---------------- pre-pass B: permute weights to (kh,kw,c) bf16 + c2 --------
__global__ void rbf_prep_kernel(const float* __restrict__ w,
                                unsigned short* __restrict__ wB,
                                float* __restrict__ c2) {
    const int o = blockIdx.x;
    const float* wr = w + (size_t)o * KDIM;
    float ss = 0.f;
    for (int k = threadIdx.x; k < KDIM; k += blockDim.x) {
        float v = wr[k];
        ss += v * v;
        const int c = k / 9, r = k % 9;
        wB[(size_t)o * KDIM + r * IN_CH + c] = f32_to_bf16_rne(v);
    }
    __shared__ float red;
    if (threadIdx.x == 0) red = 0.f;
    __syncthreads();
    atomicAdd(&red, ss);
    __syncthreads();
    if (threadIdx.x == 0) c2[o] = red;
}

// ---------------- TDM-fed bf16 WMMA GEMM + RBF epilogue ----------------
__global__ __launch_bounds__(256)
void rbf_conv_wmma_kernel(const unsigned short* __restrict__ xT,
                          const float* __restrict__ sq,
                          const unsigned short* __restrict__ wB,
                          const float* __restrict__ c2,
                          float* __restrict__ out) {
    __shared__ __align__(16) char ldsRaw[2 * ABUF_BYTES];   // A double buffer
    __shared__ float ldsW2[MTILE];
    unsigned short* ldsA = (unsigned short*)ldsRaw;
    float*          ldsOut = (float*)ldsRaw;                // epilogue alias

    const int t     = threadIdx.x;
    const int lane  = t & 31;
    const int wave  = t >> 5;
    const int waveM = wave >> 1;           // 0..3 -> 32-row strip
    const int waveN = wave & 1;            // 0..1 -> 64-col strip

    const int b   = blockIdx.z & 15;
    const int nb  = blockIdx.z >> 4;
    const int wo0 = blockIdx.x * WTILE;
    const int ho0 = blockIdx.y * HTILE;
    const int nBase = nb * NTILE;

    const unsigned short* xTb = xT + (size_t)b * HW * HW * IN_CH;
    const unsigned ldsA_off = (unsigned)(size_t)(void*)ldsRaw;

    // prologue: kick tap 0 DMA; overlap w2 (3x3 box filter of sq) with it
    if (wave == 0) tdm_load_a(xTb, ldsA_off, ho0, wo0);

    if (t < MTILE) {
        const int hl = t >> 4, wl = t & 15;
        const float* sqb = sq + (size_t)b * HW * HW;
        float s = 0.f;
#pragma unroll
        for (int kh = 0; kh < 3; ++kh) {
#pragma unroll
            for (int kw = 0; kw < 3; ++kw) {
                int hh = ho0 + hl + kh; if (hh > HW - 1) hh = HW - 1;
                int ww = wo0 + wl + kw; if (ww > HW - 1) ww = HW - 1;
                s += sqb[hh * HW + ww];
            }
        }
        ldsW2[t] = s;
    }

    if (wave == 0) __builtin_amdgcn_s_wait_tensorcnt(0);
    __syncthreads();

    v8f acc[2][4] = {};

    for (int r = 0; r < 9; ++r) {
        const int cur = r & 1;
        if (r < 8 && wave == 0) {
            const int rn = r + 1;
            tdm_load_a(xTb, ldsA_off + (unsigned)((rn & 1) * ABUF_BYTES),
                       ho0 + rn / 3, wo0 + rn % 3);
        }
        const unsigned short* A = ldsA + cur * (MTILE * PITCHA);
#pragma unroll
        for (int k0 = 0; k0 < 64; k0 += 32) {
            v16bf a0 = load_a_frag(A, waveM * 32,      k0, lane);
            v16bf a1 = load_a_frag(A, waveM * 32 + 16, k0, lane);
            const int kp = r * 64 + k0;
#pragma unroll
            for (int s = 0; s < 4; ++s) {
                v16bf bf = load_b_frag(wB, nBase + waveN * 64 + s * 16, kp, lane);
                acc[0][s] = __builtin_amdgcn_wmma_f32_16x16x32_bf16(
                    false, a0, false, bf, (short)0, acc[0][s], false, false);
                acc[1][s] = __builtin_amdgcn_wmma_f32_16x16x32_bf16(
                    false, a1, false, bf, (short)0, acc[1][s], false, false);
            }
        }
        if (r < 8 && wave == 0) __builtin_amdgcn_s_wait_tensorcnt(0);
        __syncthreads();
    }

    // -------- epilogue: RBF activation + LDS transpose + coalesced store ----
    for (int h = 0; h < 2; ++h) {
        __syncthreads();
        if (waveN == h) {
#pragma unroll
            for (int i = 0; i < 2; ++i) {
#pragma unroll
                for (int s = 0; s < 4; ++s) {
                    const int n_l = s * 16 + (lane & 15);
                    const float cc = c2[nBase + h * 64 + n_l];
#pragma unroll
                    for (int rr = 0; rr < 8; ++rr) {
                        const int m_l = waveM * 32 + i * 16 + rr + (lane >> 4) * 8;
                        float d2 = ldsW2[m_l] + cc - 2.0f * acc[i][s][rr];
                        d2 = fmaxf(d2, 1e-12f);
                        ldsOut[n_l * PITCHO + m_l] = __expf(-d2 * 0.125f);
                    }
                }
            }
        }
        __syncthreads();

        float* outB = out + ((size_t)b * OUT_CH + nBase + h * 64) * P;
#pragma unroll
        for (int i = 0; i < 8; ++i) {
            const int o_l = i * 8 + (t >> 5);        // 0..63
            const int hl  = lane >> 2;               // 0..7
            const int w4  = (lane & 3) * 4;          // 0,4,8,12
            const int ho  = ho0 + hl;
            if (ho < HO) {
                float* dst = outB + (size_t)o_l * P + ho * HO + wo0 + w4;
                const float* srcL = ldsOut + o_l * PITCHO + hl * WTILE + w4;
#pragma unroll
                for (int j = 0; j < 4; ++j)
                    if (wo0 + w4 + j < HO) dst[j] = srcL[j];
            }
        }
    }
}

extern "C" void kernel_launch(void* const* d_in, const int* in_sizes, int n_in,
                              void* d_out, int out_size, void* d_ws, size_t ws_size,
                              hipStream_t stream) {
    const float* x = (const float*)d_in[0];      // [16,64,56,56] f32
    const float* w = (const float*)d_in[1];      // [256,576] f32
    float* out = (float*)d_out;                  // [16,256,54,54] f32

    // workspace layout
    unsigned short* wB = (unsigned short*)d_ws;                         // 294912 B
    float* c2 = (float*)((char*)d_ws + (size_t)OUT_CH * KDIM * 2);      //   1024 B
    char* p = (char*)d_ws + (size_t)OUT_CH * KDIM * 2 + 1024;
    unsigned short* xT = (unsigned short*)p;                            // 6422528 B
    float* sq = (float*)(p + (size_t)BATCH * HW * HW * IN_CH * 2);      //  200704 B

    rbf_xt_kernel<<<(BATCH * HW * HW) / 256, 256, 0, stream>>>(
        x, (unsigned int*)xT, sq);
    rbf_prep_kernel<<<OUT_CH, 64, 0, stream>>>(w, wB, c2);

    dim3 grid((HO + WTILE - 1) / WTILE,          // 4 wo tiles
              (HO + HTILE - 1) / HTILE,          // 7 ho tiles
              BATCH * (OUT_CH / NTILE));         // 32
    rbf_conv_wmma_kernel<<<grid, 256, 0, stream>>>(xT, sq, wB, c2, out);
}
